// MultiHeadAttention_3375844294887
// MI455X (gfx1250) — matrix operand order repaired and checked by reading, not compiled, not run on hardware
//
#include <hip/hip_runtime.h>

// ---------- types for WMMA ----------
typedef __attribute__((ext_vector_type(16))) __bf16 v16bf;
typedef __attribute__((ext_vector_type(8)))  float  v8f;

union FragBF { v16bf v; unsigned u[8]; };
union FragF  { v8f  v; float    f[8]; };

static __device__ inline unsigned short f2bf(float f) {
    unsigned u = __float_as_uint(f);
    unsigned r = 0x7FFFu + ((u >> 16) & 1u);
    return (unsigned short)((u + r) >> 16);
}

// ---------- CDNA5 async global->LDS copy (ASYNCcnt path) ----------
static __device__ inline unsigned ldsAddr(const void* p) {
    // generic pointer to LDS: low 32 bits are the LDS byte address (ISA 10.2)
    return (unsigned)(unsigned long long)p;
}
static __device__ inline void asyncLoadB128(void* lds, const void* gaddr) {
    asm volatile("global_load_async_to_lds_b128 %0, %1, off"
                 :: "v"(ldsAddr(lds)), "v"((unsigned long long)gaddr)
                 : "memory");
}
// in-order completion: waiting <=N leaves the most recent N loads in flight
static __device__ inline void waitAsync0() {
    asm volatile("s_wait_asynccnt 0x0" ::: "memory");
}
static __device__ inline void waitAsync2() {
    asm volatile("s_wait_asynccnt 0x2" ::: "memory");
}
static __device__ inline void waitAsync3() {
    asm volatile("s_wait_asynccnt 0x3" ::: "memory");
}

// ---------- constants ----------
#define B_   4
#define S_   2048
#define D_   1024
#define H_   16
#define DK_  64
#define M_   (B_ * S_)          // 8192
#define NT_  1024               // GEMM N
#define KT_  1024               // GEMM K

// ---------- elementwise f32 -> bf16 ----------
__global__ void cvt_f32_bf16(const float* __restrict__ src,
                             unsigned short* __restrict__ dst, int n) {
    int i = blockIdx.x * blockDim.x + threadIdx.x;
    if (i < n) dst[i] = f2bf(src[i]);
}

// ---------- weights -> pair-swizzled fragment-ready bf16 layout ----------
// Word i (u32 = bf16 pair {K=k, K=k+1}) at:
//   i = nb*32768 + kt*1024 + col*16 + half*8 + j
//   k = kt*32 + half*16 + 2*j ; n = nb*64 + col
// Per (nb,kt) tile = 1024 contiguous words (4KB): straight copy to LDS, and each
// WMMA lane's 8 B-fragment words are contiguous (two ds_load_b128, no packing).
__global__ void cvt_w_swz(const float* __restrict__ W,
                          unsigned* __restrict__ out, int qkv) {
    int i = blockIdx.x * blockDim.x + threadIdx.x;   // 524288 words
    if (i >= (KT_ / 2) * NT_) return;
    int widx = i & 15;
    int col  = (i >> 4) & 63;
    int kt   = (i >> 10) & 31;
    int nb   = i >> 15;
    int half = widx >> 3, j = widx & 7;
    int k = kt * 32 + half * 16 + 2 * j;
    float lo, hi;
    if (qkv) {   // W [H,D,64]: B[k=d][n=h*64+dk] = W[h][d][dk]; nb==h, col==dk
        lo = W[nb * (D_ * DK_) + k * DK_ + col];
        hi = W[nb * (D_ * DK_) + (k + 1) * DK_ + col];
    } else {     // Wo [1024,1024] row-major: B[k][n] = Wo[k][n]
        int n = nb * 64 + col;
        lo = W[k * NT_ + n];
        hi = W[(k + 1) * NT_ + n];
    }
    out[i] = (unsigned)f2bf(lo) | ((unsigned)f2bf(hi) << 16);
}

// ---------- bf16 GEMM: C[8192,1024] = A[8192,1024] * B[1024,1024] ----------
// Block tile 128x64, K-step 32, 8 waves each owning 32x32 (2x2 WMMA tiles).
// Double-buffered LDS; next tile's async-to-LDS loads fly behind current WMMAs.
__global__ __launch_bounds__(256) void gemm_bf16_kernel(
    const unsigned short* __restrict__ A, const unsigned* __restrict__ Bswz,
    float* __restrict__ outF, unsigned short* __restrict__ outQ) {
    __shared__ __align__(16) unsigned short Atile[2][128 * 32];   // 16 KB
    __shared__ __align__(16) unsigned       Btile[2][1024];       //  8 KB

    const int tid  = threadIdx.x;
    const int lane = tid & 31;
    const int w    = tid >> 5;
    const int wm   = w & 3, wn = w >> 2;
    const int mbase = blockIdx.y * 128;
    const int nb    = blockIdx.x;            // 64-col block
    const int half  = lane >> 4, n0 = lane & 15;

    FragF acc[2][2];
    for (int a = 0; a < 2; ++a)
        for (int c = 0; c < 2; ++c)
            for (int r = 0; r < 8; ++r) acc[a][c].f[r] = 0.f;

    const int arow = tid >> 1, akoff = (tid & 1) * 16;
    const unsigned short* aSrc = A + (size_t)(mbase + arow) * KT_ + akoff;
    const unsigned*       bSrc = Bswz + (size_t)nb * 32768 + tid * 4;

    auto issueTile = [&](int kt, int buf) {
        asyncLoadB128(&Atile[buf][arow * 32 + akoff],     aSrc + kt * 32);
        asyncLoadB128(&Atile[buf][arow * 32 + akoff + 8], aSrc + kt * 32 + 8);
        asyncLoadB128(&Btile[buf][tid * 4],               bSrc + kt * 1024);
    };

    issueTile(0, 0);                          // prologue
    for (int kt = 0; kt < 32; ++kt) {
        const int cur = kt & 1;
        if (kt + 1 < 32) {                    // prefetch next tile, keep it in flight
            issueTile(kt + 1, cur ^ 1);
            waitAsync3();                     // current tile's 3 loads landed
        } else {
            waitAsync0();
        }
        __syncthreads();

        FragBF afr[2], bfr[2];
        #pragma unroll
        for (int tm = 0; tm < 2; ++tm) {      // A frag: contiguous k = half*16 + s
            int row = wm * 32 + tm * 16 + n0;
            #pragma unroll
            for (int j = 0; j < 8; ++j) {
                int k = half * 16 + j * 2;
                afr[tm].u[j] = *(const unsigned*)&Atile[cur][row * 32 + k];
            }
        }
        #pragma unroll
        for (int tn = 0; tn < 2; ++tn) {      // B frag: contiguous 8 words
            const unsigned* bp =
                &Btile[cur][(wn * 32 + tn * 16 + n0) * 16 + half * 8];
            #pragma unroll
            for (int j = 0; j < 8; ++j) bfr[tn].u[j] = bp[j];
        }
        #pragma unroll
        for (int tm = 0; tm < 2; ++tm)
            #pragma unroll
            for (int tn = 0; tn < 2; ++tn)
                acc[tm][tn].v = __builtin_amdgcn_wmma_f32_16x16x32_bf16(
                    false, afr[tm].v, false, bfr[tn].v,
                    (short)0, acc[tm][tn].v, false, false);
        __syncthreads();                      // all reads of `cur` done
    }

    // epilogue (C/D layout: vgpr r -> M=(lane/16)*8+r, N=lane%16)
    const int mh = half * 8;
    if (outF) {
        for (int tm = 0; tm < 2; ++tm)
            for (int tn = 0; tn < 2; ++tn)
                for (int r = 0; r < 8; ++r) {
                    int gm = mbase + wm * 32 + tm * 16 + mh + r;
                    int gn = nb * 64 + wn * 32 + tn * 16 + n0;
                    outF[(size_t)gm * NT_ + gn] = acc[tm][tn].f[r];
                }
    } else {
        for (int tm = 0; tm < 2; ++tm)
            for (int tn = 0; tn < 2; ++tn)
                for (int r = 0; r < 8; ++r) {
                    int gm = mbase + wm * 32 + tm * 16 + mh + r;
                    int gn = nb * 64 + wn * 32 + tn * 16 + n0;
                    int bb = gm >> 11, s = gm & (S_ - 1);
                    int h = gn >> 6, k2 = gn & 63;
                    outQ[(((size_t)(bb * H_ + h) * S_ + s) << 6) + k2] =
                        f2bf(acc[tm][tn].f[r]);
                }
    }
}

// ---------- flash attention: one block per (b, h, 128 query rows) ----------
// Q,K,V in [B,H,S,64] bf16; heads out in [B,S,H*64] bf16 (concat layout).
__global__ __launch_bounds__(256) void attn_kernel(
    const unsigned short* __restrict__ Q, const unsigned short* __restrict__ Km,
    const unsigned short* __restrict__ V, unsigned short* __restrict__ heads) {
    __shared__ __align__(16) unsigned short Klds[2][64 * 64];   // [key][d], 16 KB
    __shared__ __align__(16) unsigned       VldsW[2][64 * 32];  // [d][keypair], 16 KB
    __shared__ __align__(16) unsigned short Plds[8][16 * 64];   // 16 KB

    const int tid  = threadIdx.x, lane = tid & 31, w = tid >> 5;
    const int h    = blockIdx.y, b = blockIdx.z;
    const int bh   = b * H_ + h;
    const int qbase = blockIdx.x * 128;
    const unsigned short* Qh = Q  + (size_t)bh * S_ * DK_;
    const unsigned short* Kh = Km + (size_t)bh * S_ * DK_;
    const unsigned short* Vh = V  + (size_t)bh * S_ * DK_;

    const int half = lane >> 4;
    const int n0   = lane & 15;

    // Q A-fragments (loop-invariant)
    FragBF aq[2];
    {
        int row = qbase + w * 16 + n0;
        #pragma unroll
        for (int ks = 0; ks < 2; ++ks)
            #pragma unroll
            for (int j = 0; j < 8; ++j) {
                int k = ks * 32 + half * 16 + j * 2;
                aq[ks].u[j] = *(const unsigned*)&Qh[(size_t)row * DK_ + k];
            }
    }

    float m[8], l[8];
    for (int r = 0; r < 8; ++r) { m[r] = -1e30f; l[r] = 0.f; }
    FragF accO[4];
    for (int t = 0; t < 4; ++t)
        for (int r = 0; r < 8; ++r) accO[t].f[r] = 0.f;

    const float scale = 0.125f;   // 1/sqrt(64)

    const int krow = tid >> 2, kcoff = (tid & 3) * 16;
    const int rp = tid >> 3, doff = (tid & 7) * 8;   // keypair, d-range

    auto issueK = [&](int kb2, int buf) {
        asyncLoadB128(&Klds[buf][krow * 64 + kcoff],
                      &Kh[(size_t)(kb2 + krow) * 64 + kcoff]);
        asyncLoadB128(&Klds[buf][krow * 64 + kcoff + 8],
                      &Kh[(size_t)(kb2 + krow) * 64 + kcoff + 8]);
    };
    auto loadV = [&](int kb2, uint4& va, uint4& vb) {
        va = *(const uint4*)&Vh[(size_t)(kb2 + 2 * rp) * 64 + doff];
        vb = *(const uint4*)&Vh[(size_t)(kb2 + 2 * rp + 1) * 64 + doff];
    };
    auto storeV = [&](int buf, const uint4& va, const uint4& vb) {
        const unsigned short* ap = (const unsigned short*)&va;
        const unsigned short* bp = (const unsigned short*)&vb;
        #pragma unroll
        for (int i = 0; i < 8; ++i)
            VldsW[buf][(doff + i) * 32 + rp] =
                (unsigned)ap[i] | ((unsigned)bp[i] << 16);
    };

    // prologue: stage key block 0
    issueK(0, 0);
    {
        uint4 va, vb;
        loadV(0, va, vb);
        storeV(0, va, vb);
    }

    for (int ib = 0; ib < 32; ++ib) {
        const int cur = ib & 1;
        const bool more = (ib + 1 < 32);
        uint4 va, vb;
        if (more) {                       // prefetch next K (async) + V (regs)
            issueK((ib + 1) * 64, cur ^ 1);
            loadV((ib + 1) * 64, va, vb);
            waitAsync2();                 // current K tile's 2 loads landed
        } else {
            waitAsync0();
        }
        __syncthreads();

        // S = scale * Q K^T (16x64 per wave)
        FragF sc[4];
        for (int t = 0; t < 4; ++t) {
            for (int r = 0; r < 8; ++r) sc[t].f[r] = 0.f;
            #pragma unroll
            for (int ks = 0; ks < 2; ++ks) {
                FragBF bK;
                int col = t * 16 + n0;
                #pragma unroll
                for (int j = 0; j < 8; ++j)
                    bK.u[j] = *(const unsigned*)
                        &Klds[cur][col * 64 + ks * 32 + half * 16 + 2 * j];
                sc[t].v = __builtin_amdgcn_wmma_f32_16x16x32_bf16(
                    false, aq[ks].v, false, bK.v, (short)0, sc[t].v, false, false);
            }
            for (int r = 0; r < 8; ++r) sc[t].f[r] *= scale;
        }

        // online softmax (rows live in 16-lane half-waves)
        float mnew[8], scl[8];
        for (int r = 0; r < 8; ++r) {
            float v = fmaxf(fmaxf(sc[0].f[r], sc[1].f[r]),
                            fmaxf(sc[2].f[r], sc[3].f[r]));
            for (int mask = 8; mask >= 1; mask >>= 1)
                v = fmaxf(v, __shfl_xor(v, mask, 32));
            mnew[r] = fmaxf(m[r], v);
            scl[r]  = __expf(m[r] - mnew[r]);
            m[r]    = mnew[r];
        }

        // P = exp(S - mnew); rowsum; stash P per-wave for D->A layout swap
        {
            float rs[8];
            for (int r = 0; r < 8; ++r) rs[r] = 0.f;
            for (int t = 0; t < 4; ++t)
                for (int r = 0; r < 8; ++r) {
                    float p = __expf(sc[t].f[r] - mnew[r]);
                    rs[r] += p;
                    Plds[w][(half * 8 + r) * 64 + t * 16 + n0] = f2bf(p);
                }
            for (int r = 0; r < 8; ++r) {
                float v = rs[r];
                for (int mask = 8; mask >= 1; mask >>= 1)
                    v += __shfl_xor(v, mask, 32);
                l[r] = l[r] * scl[r] + v;
            }
        }

        // rescale O then O += P*V
        for (int t = 0; t < 4; ++t)
            for (int r = 0; r < 8; ++r) accO[t].f[r] *= scl[r];

        FragBF ap2[2];   // P as A-matrix (same-wave LDS ops are in-order)
        #pragma unroll
        for (int ks = 0; ks < 2; ++ks)
            #pragma unroll
            for (int j = 0; j < 8; ++j) {
                int k = ks * 32 + half * 16 + j * 2;
                ap2[ks].u[j] = *(const unsigned*)&Plds[w][n0 * 64 + k];
            }
        for (int t = 0; t < 4; ++t) {
            #pragma unroll
            for (int ks = 0; ks < 2; ++ks) {
                FragBF bV;   // contiguous 8 words from packed V
                const unsigned* vp =
                    &VldsW[cur][(t * 16 + n0) * 32 + ks * 16 + half * 8];
                #pragma unroll
                for (int j = 0; j < 8; ++j) bV.u[j] = vp[j];
                accO[t].v = __builtin_amdgcn_wmma_f32_16x16x32_bf16(
                    false, ap2[ks].v, false, bV.v, (short)0, accO[t].v,
                    false, false);
            }
        }

        if (more) storeV(cur ^ 1, va, vb);   // stage next V after compute
        __syncthreads();                     // all reads of `cur` done
    }

    // epilogue: O / l -> heads[b, s, h*64+d]
    for (int t = 0; t < 4; ++t)
        for (int r = 0; r < 8; ++r) {
            float o = accO[t].f[r] / l[r];
            int srow = qbase + w * 16 + half * 8 + r;
            int d    = t * 16 + n0;
            heads[((size_t)(b * S_ + srow) * (H_ * DK_)) + h * DK_ + d] = f2bf(o);
        }
}

// ---------- host launcher ----------
extern "C" void kernel_launch(void* const* d_in, const int* in_sizes, int n_in,
                              void* d_out, int out_size, void* d_ws, size_t ws_size,
                              hipStream_t stream) {
    const float* x  = (const float*)d_in[0];
    const float* Wq = (const float*)d_in[1];
    const float* Wk = (const float*)d_in[2];
    const float* Wv = (const float*)d_in[3];
    const float* Wo = (const float*)d_in[4];

    char* ws = (char*)d_ws;
    size_t off = 0;
    auto alloc = [&](size_t bytes) -> void* {
        void* p = ws + off;
        off = (off + bytes + 255) & ~(size_t)255;
        return p;
    };
    unsigned short* xb     = (unsigned short*)alloc((size_t)M_ * D_ * 2);
    unsigned*       Wqb    = (unsigned*)alloc((size_t)(KT_ / 2) * NT_ * 4);
    unsigned*       Wkb    = (unsigned*)alloc((size_t)(KT_ / 2) * NT_ * 4);
    unsigned*       Wvb    = (unsigned*)alloc((size_t)(KT_ / 2) * NT_ * 4);
    unsigned*       Wob    = (unsigned*)alloc((size_t)(KT_ / 2) * NT_ * 4);
    unsigned short* Qb     = (unsigned short*)alloc((size_t)B_ * H_ * S_ * DK_ * 2);
    unsigned short* Kb     = (unsigned short*)alloc((size_t)B_ * H_ * S_ * DK_ * 2);
    unsigned short* Vb     = (unsigned short*)alloc((size_t)B_ * H_ * S_ * DK_ * 2);
    unsigned short* headsb = (unsigned short*)alloc((size_t)M_ * H_ * DK_ * 2);

    const int nx = M_ * D_;                 // 8388608
    const int nwords = (KT_ / 2) * NT_;     // 524288
    cvt_f32_bf16<<<(nx + 255) / 256, 256, 0, stream>>>(x, xb, nx);
    cvt_w_swz<<<(nwords + 255) / 256, 256, 0, stream>>>(Wq, Wqb, 1);
    cvt_w_swz<<<(nwords + 255) / 256, 256, 0, stream>>>(Wk, Wkb, 1);
    cvt_w_swz<<<(nwords + 255) / 256, 256, 0, stream>>>(Wv, Wvb, 1);
    cvt_w_swz<<<(nwords + 255) / 256, 256, 0, stream>>>(Wo, Wob, 0);

    dim3 gg(NT_ / 64, M_ / 128);            // (16, 64)
    gemm_bf16_kernel<<<gg, 256, 0, stream>>>(xb, Wqb, nullptr, Qb);
    gemm_bf16_kernel<<<gg, 256, 0, stream>>>(xb, Wkb, nullptr, Kb);
    gemm_bf16_kernel<<<gg, 256, 0, stream>>>(xb, Wvb, nullptr, Vb);

    attn_kernel<<<dim3(S_ / 128, H_, B_), 256, 0, stream>>>(Qb, Kb, Vb, headsb);

    gemm_bf16_kernel<<<gg, 256, 0, stream>>>(headsb, Wob, (float*)d_out, nullptr);
}
